// MutiQueryAttention_49039936586404
// MI455X (gfx1250) — matrix operand order repaired and checked by reading, not compiled, not run on hardware
//
#include <hip/hip_runtime.h>

// ---------------------------------------------------------------------------
// MQA forward for MI455X (gfx1250): bf16 WMMA pipeline, fp32 accumulate.
// B=2, S=2048, HID=2048, NH=16, HD=128, single shared KV head.
// GEMMs use 32x64 per-wave register tiles (2x A-frag x 4x B-frag reuse).
// ---------------------------------------------------------------------------

#define BATCH 2
#define SEQ   2048
#define HIDN  2048
#define NHEAD 16
#define HDIM  128

typedef __attribute__((ext_vector_type(16))) __bf16 v16bf;
typedef __attribute__((ext_vector_type(8)))  __bf16 bf16x8;
typedef __attribute__((ext_vector_type(8)))  float  v8f;

__device__ __forceinline__ v16bf cat8(bf16x8 lo, bf16x8 hi) {
  return __builtin_shufflevector(lo, hi, 0,1,2,3,4,5,6,7,8,9,10,11,12,13,14,15);
}

// A fragment (16x32 bf16, MxK): lane = (m = lane&15, h = lane>>4)
// elems 0..7  = A[m][k0 + h*8 + j]        (16B contiguous)
// elems 8..15 = A[m][k0 + 16 + h*8 + j]   (16B contiguous)
__device__ __forceinline__ v16bf load_a_frag(const __bf16* __restrict__ row, int k0, int lane) {
  const int h = lane >> 4;
  bf16x8 lo = *(const bf16x8*)(row + k0 + h * 8);
  bf16x8 hi = *(const bf16x8*)(row + k0 + 16 + h * 8);
  return cat8(lo, hi);
}

// B fragment (32x16 bf16, KxN): lane = (n = lane&15, h = lane>>4)
// elems 0..7 = B[k0 + h*8 + j][n], elems 8..15 = B[k0 + 16 + h*8 + j][n];
// with W stored row-major [n][k] these are two 16B runs of row n of W.
__device__ __forceinline__ v16bf load_b_frag(const __bf16* __restrict__ wrow, int k0, int lane) {
  const int h = lane >> 4;
  bf16x8 lo = *(const bf16x8*)(wrow + k0 + h * 8);
  bf16x8 hi = *(const bf16x8*)(wrow + k0 + 16 + h * 8);
  return cat8(lo, hi);
}

__device__ __forceinline__ v8f wmma_bf16(v16bf a, v16bf b, v8f c) {
  return __builtin_amdgcn_wmma_f32_16x16x32_bf16(false, a, false, b, (short)0, c, false, false);
}

// ---------------------------------------------------------------------------
// fp32 -> bf16 conversion
// ---------------------------------------------------------------------------
__global__ __launch_bounds__(256) void cvt_f32_bf16(const float* __restrict__ in,
                                                    __bf16* __restrict__ out, int n) {
  int i = (blockIdx.x * blockDim.x + threadIdx.x) * 4;
  if (i + 3 < n) {
    float4 f = *(const float4*)(in + i);
    out[i + 0] = (__bf16)f.x;
    out[i + 1] = (__bf16)f.y;
    out[i + 2] = (__bf16)f.z;
    out[i + 3] = (__bf16)f.w;
  } else {
    for (; i < n; ++i) out[i] = (__bf16)in[i];
  }
}

// ---------------------------------------------------------------------------
// WMMA GEMM: C[m,n] = sum_k A[m,k] * W[n,k] + bias[n]
// A: bf16 [M][K] row-major; W: bf16 [N][K] row-major.
// One wave computes a 32x64 tile: 2 A fragments x 4 B fragments -> 8 WMMAs
// per 32-wide k-step with only 12 b128 loads (B reused across both M rows).
// STORE_MODE: 0 = bf16 [M][N], 1 = bf16 transposed per-batch [B][N][S], 2 = f32 [M][N]
// ---------------------------------------------------------------------------
template <int STORE_MODE>
__global__ __launch_bounds__(256) void wmma_gemm(const __bf16* __restrict__ A,
                                                 const __bf16* __restrict__ W,
                                                 const float* __restrict__ bias,
                                                 void* __restrict__ Cout,
                                                 int M, int N, int K) {
  const int wave = blockIdx.x * (blockDim.x >> 5) + (threadIdx.x >> 5);
  const int ntiles = N >> 6;                 // 64-wide N tiles
  const int mt = wave / ntiles;
  const int nt = wave % ntiles;
  if (mt >= (M >> 5)) return;
  const int m0 = mt << 5;                    // 32-high M tiles
  const int n0 = nt << 6;
  const int lane = threadIdx.x & 31;
  const int lr = lane & 15;
  const int lh = lane >> 4;

  const __bf16* arow0 = A + (size_t)(m0 + lr) * K;
  const __bf16* arow1 = arow0 + (size_t)16 * K;
  const __bf16* wrow0 = W + (size_t)(n0 + lr) * K;

  v8f acc[2][4] = {{v8f{}, v8f{}, v8f{}, v8f{}}, {v8f{}, v8f{}, v8f{}, v8f{}}};
  for (int k0 = 0; k0 < K; k0 += 32) {
    v16bf a0 = load_a_frag(arow0, k0, lane);
    v16bf a1 = load_a_frag(arow1, k0, lane);
#pragma unroll
    for (int t = 0; t < 4; ++t) {
      v16bf b = load_b_frag(wrow0 + (size_t)(t * 16) * K, k0, lane);
      acc[0][t] = wmma_bf16(a0, b, acc[0][t]);
      acc[1][t] = wmma_bf16(a1, b, acc[1][t]);
    }
  }

#pragma unroll
  for (int mi = 0; mi < 2; ++mi) {
#pragma unroll
    for (int t = 0; t < 4; ++t) {
      const int n = n0 + t * 16 + lr;
      const float bv = bias ? bias[n] : 0.0f;
#pragma unroll
      for (int r = 0; r < 8; ++r) {
        const int m = m0 + mi * 16 + r + 8 * lh;
        const float val = acc[mi][t][r] + bv;
        if (STORE_MODE == 0) {
          ((__bf16*)Cout)[(size_t)m * N + n] = (__bf16)val;
        } else if (STORE_MODE == 1) {
          const int b = m / SEQ, s = m % SEQ;
          ((__bf16*)Cout)[((size_t)b * N + n) * SEQ + s] = (__bf16)val;
        } else {
          ((float*)Cout)[(size_t)m * N + n] = val;
        }
      }
    }
  }
}

// ---------------------------------------------------------------------------
// Flash MQA attention. One wave handles one (b, h, 16-query tile).
// Q: bf16 [B][S][HID], K: bf16 [B][S][HD], Vt: bf16 [B][HD][S],
// mask: f32 [B][1][S][S], OutT: bf16 flat [B][NH][HD][S] (== reference's
// transpose(0,1,3,2).reshape(B,S,HID) view).
// ---------------------------------------------------------------------------
__global__ __launch_bounds__(128) void mqa_flash(const __bf16* __restrict__ Q,
                                                 const __bf16* __restrict__ Km,
                                                 const __bf16* __restrict__ Vt,
                                                 const float* __restrict__ mask,
                                                 __bf16* __restrict__ OutT) {
  __shared__ __bf16 plds[4][16 * 32];      // per-wave P staging tile
  const int wslot = threadIdx.x >> 5;
  const int lane = threadIdx.x & 31;
  const int wave = blockIdx.x * 4 + wslot;

  const int QT = SEQ / 16;                  // 128 query tiles per (b,h)
  const int qt = wave % QT;
  const int bh = wave / QT;
  const int h = bh % NHEAD;
  const int b = bh / NHEAD;
  const int q0 = qt * 16;
  const int lr = lane & 15;
  const int lh = lane >> 4;

  // Preload Q fragments for all 4 K-steps over d (HD=128 = 4 x 32)
  const __bf16* qrow = Q + ((size_t)b * SEQ + (q0 + lr)) * HIDN + (size_t)h * HDIM;
  v16bf qf[4];
#pragma unroll
  for (int t = 0; t < 4; ++t) qf[t] = load_a_frag(qrow, t * 32, lane);

  const __bf16* kbase = Km + (size_t)b * SEQ * HDIM;
  const __bf16* vbase = Vt + (size_t)b * HDIM * SEQ;
  const float* mbase = mask + (size_t)b * SEQ * SEQ + (size_t)q0 * SEQ;

  v8f o[8] = {v8f{}, v8f{}, v8f{}, v8f{}, v8f{}, v8f{}, v8f{}, v8f{}};
  float mrow[8], lrow[8];
#pragma unroll
  for (int r = 0; r < 8; ++r) { mrow[r] = -__builtin_inff(); lrow[r] = 0.0f; }

  const float scale = 0.08838834764831845f;  // 1/sqrt(128)
  __bf16* pl = plds[wslot];

  for (int j0 = 0; j0 < SEQ; j0 += 32) {
    // ---- scores: 16x32 block = two 16x16 WMMA columns, K over d ----
    v8f s0 = v8f{}, s1 = v8f{};
#pragma unroll
    for (int t = 0; t < 4; ++t) {
      v16bf kb0 = load_b_frag(kbase + (size_t)(j0 + lr) * HDIM, t * 32, lane);
      s0 = wmma_bf16(qf[t], kb0, s0);
      v16bf kb1 = load_b_frag(kbase + (size_t)(j0 + 16 + lr) * HDIM, t * 32, lane);
      s1 = wmma_bf16(qf[t], kb1, s1);
    }

    // ---- scale + (faithful) mask * -1e-9 ----
#pragma unroll
    for (int r = 0; r < 8; ++r) {
      const int i = r + 8 * lh;            // query row within tile
      const int jc = j0 + lr;
      const float m0v = mbase[(size_t)i * SEQ + jc];
      const float m1v = mbase[(size_t)i * SEQ + jc + 16];
      s0[r] = s0[r] * scale + m0v * (-1e-9f);
      s1[r] = s1[r] * scale + m1v * (-1e-9f);
    }

    // ---- online softmax (row = fixed vgpr index across a 16-lane half) ----
#pragma unroll
    for (int r = 0; r < 8; ++r) {
      float bm = fmaxf(s0[r], s1[r]);
      bm = fmaxf(bm, __shfl_xor(bm, 1));
      bm = fmaxf(bm, __shfl_xor(bm, 2));
      bm = fmaxf(bm, __shfl_xor(bm, 4));
      bm = fmaxf(bm, __shfl_xor(bm, 8));
      const float mn = fmaxf(mrow[r], bm);
      const float alpha = __expf(mrow[r] - mn);
      mrow[r] = mn;
      const float p0 = __expf(s0[r] - mn);
      const float p1 = __expf(s1[r] - mn);
      s0[r] = p0;
      s1[r] = p1;
      float rs = p0 + p1;
      rs += __shfl_xor(rs, 1);
      rs += __shfl_xor(rs, 2);
      rs += __shfl_xor(rs, 4);
      rs += __shfl_xor(rs, 8);
      lrow[r] = lrow[r] * alpha + rs;
#pragma unroll
      for (int t = 0; t < 8; ++t) o[t][r] *= alpha;
    }

    // ---- P (C-layout) -> LDS -> A-layout fragment (wave-private, no barrier) ----
#pragma unroll
    for (int r = 0; r < 8; ++r) {
      pl[(r + 8 * lh) * 32 + lr] = (__bf16)s0[r];
      pl[(r + 8 * lh) * 32 + 16 + lr] = (__bf16)s1[r];
    }
    asm volatile("s_wait_dscnt 0x0" ::: "memory");   // LDS stores -> loads RAW
    v16bf pf;
    {
      const __bf16* pp = pl + lr * 32;
      bf16x8 plo = *(const bf16x8*)(pp + lh * 8);
      bf16x8 phi = *(const bf16x8*)(pp + 16 + lh * 8);
      pf = cat8(plo, phi);
    }
    asm volatile("s_wait_dscnt 0x0" ::: "memory");   // loads done before next overwrite

    // ---- out += P @ V  (Vt gives contiguous per-lane K-runs) ----
#pragma unroll
    for (int t = 0; t < 8; ++t) {
      v16bf vf = load_b_frag(vbase + (size_t)(t * 16 + lr) * SEQ, j0, lane);
      o[t] = wmma_bf16(pf, vf, o[t]);
    }
  }

  // ---- normalize, store in [B][NH][HD][S] flat order ----
  __bf16* obase = OutT + ((size_t)b * NHEAD + h) * HDIM * SEQ;
#pragma unroll
  for (int r = 0; r < 8; ++r) {
    const float inv = 1.0f / lrow[r];
    const int s = q0 + r + 8 * lh;
#pragma unroll
    for (int t = 0; t < 8; ++t) {
      const int d = t * 16 + lr;
      obase[(size_t)d * SEQ + s] = (__bf16)(o[t][r] * inv);
    }
  }
}

// ---------------------------------------------------------------------------
// Host-side launcher
// ---------------------------------------------------------------------------
extern "C" void kernel_launch(void* const* d_in, const int* in_sizes, int n_in,
                              void* d_out, int out_size, void* d_ws, size_t ws_size,
                              hipStream_t stream) {
  (void)in_sizes; (void)n_in; (void)out_size; (void)ws_size;

  const float* hidden = (const float*)d_in[0];
  const float* amask  = (const float*)d_in[1];
  const float* Wq = (const float*)d_in[2];
  const float* bq = (const float*)d_in[3];
  const float* Wk = (const float*)d_in[4];
  const float* bk = (const float*)d_in[5];
  const float* Wv = (const float*)d_in[6];
  const float* bv = (const float*)d_in[7];
  const float* Wo = (const float*)d_in[8];
  const float* bo = (const float*)d_in[9];
  float* out = (float*)d_out;

  const int M = BATCH * SEQ;  // 4096

  // Carve workspace (bf16 staging, 256B aligned slices)
  size_t off = 0;
  auto carve = [&](size_t elems) {
    __bf16* p = (__bf16*)((char*)d_ws + off);
    off = (off + elems * sizeof(__bf16) + 255) & ~(size_t)255;
    return p;
  };
  __bf16* hid_bf = carve((size_t)M * HIDN);
  __bf16* wq_bf  = carve((size_t)HIDN * HIDN);
  __bf16* wk_bf  = carve((size_t)HDIM * HIDN);
  __bf16* wv_bf  = carve((size_t)HDIM * HIDN);
  __bf16* wo_bf  = carve((size_t)HIDN * HIDN);
  __bf16* q_bf   = carve((size_t)M * HIDN);
  __bf16* k_bf   = carve((size_t)M * HDIM);
  __bf16* vt_bf  = carve((size_t)M * HDIM);   // [B][HD][S]
  __bf16* at_bf  = carve((size_t)M * HIDN);   // [B][NH][HD][S] flat

  auto cvt = [&](const float* src, __bf16* dst, int n) {
    int blocks = (n / 4 + 255) / 256;
    cvt_f32_bf16<<<blocks, 256, 0, stream>>>(src, dst, n);
  };
  cvt(hidden, hid_bf, M * HIDN);
  cvt(Wq, wq_bf, HIDN * HIDN);
  cvt(Wk, wk_bf, HDIM * HIDN);
  cvt(Wv, wv_bf, HDIM * HIDN);
  cvt(Wo, wo_bf, HIDN * HIDN);

  auto gemm_blocks = [&](int Mm, int Nn) {
    int waves = (Mm >> 5) * (Nn >> 6);   // 32x64 tiles per wave
    return (waves + 7) / 8;              // 8 waves (256 threads) per block
  };

  // Q = X Wq^T + bq   -> bf16 [M][HID]
  wmma_gemm<0><<<gemm_blocks(M, HIDN), 256, 0, stream>>>(hid_bf, wq_bf, bq, q_bf, M, HIDN, HIDN);
  // K = X Wk^T + bk   -> bf16 [B][S][HD]
  wmma_gemm<0><<<gemm_blocks(M, HDIM), 256, 0, stream>>>(hid_bf, wk_bf, bk, k_bf, M, HDIM, HIDN);
  // Vt = (X Wv^T + bv)^T per batch -> bf16 [B][HD][S]
  wmma_gemm<1><<<gemm_blocks(M, HDIM), 256, 0, stream>>>(hid_bf, wv_bf, bv, vt_bf, M, HDIM, HIDN);

  // Flash MQA: B*NH*(S/16) = 4096 waves, 4 waves/block
  mqa_flash<<<(BATCH * NHEAD * (SEQ / 16)) / 4, 128, 0, stream>>>(q_bf, k_bf, vt_bf, amask, at_bf);

  // Y = attn(view [B*S][HID]) Wo^T + bo -> f32 d_out
  wmma_gemm<2><<<gemm_blocks(M, HIDN), 256, 0, stream>>>(at_bf, wo_bf, bo, out, M, HIDN, HIDN);
}
